// SheafConvLayer_8924942041558
// MI455X (gfx1250) — compile-verified, matrix operand order checked by reading, not run on hardware
//
#include <hip/hip_runtime.h>

typedef __attribute__((ext_vector_type(2)))  float    v2f;
typedef __attribute__((ext_vector_type(8)))  float    v8f;
typedef __attribute__((ext_vector_type(16))) _Float16 v16h;

// Compute y = W (16x16 row-major f32) * xvec (16 f32) via WMMA in the
// TRANSPOSED formulation:  D = A x B with A = x^T replicated in all 16 rows,
// B = W^T.  Then D[m][n] = sum_k x[k] * W[n][k] = y[n] for every row m, so
// after the WMMA each lane holds y[lane & 15] in ALL of its accumulator VGPRs:
// no cross-lane redistribution is needed for a row-contiguous 64B store.
__device__ __forceinline__ float matvec16_wmma(const float* __restrict__ W,
                                               const float* __restrict__ xv,
                                               int lane) {
  const int half = (lane >> 4) & 1;   // 0: lanes 0-15, 1: lanes 16-31
  const int row  = lane & 15;

#if __has_builtin(__builtin_amdgcn_wmma_f32_16x16x4_f32)
  // Exact f32 path: 4 chained V_WMMA_F32_16X16X4_F32 over K chunks of 4.
  // A 16x4 f32 layout: lane<16 -> {K=0,K=1}, lane>=16 -> {K=2,K=3} (M=lane&15).
  // B 4x16 f32 layout mirrors it with N=lane&15.
  const int koff = half * 2;
  v8f c = {};
#pragma unroll
  for (int ch = 0; ch < 4; ++ch) {
    const int kb = 4 * ch + koff;
    v2f a = *(const v2f*)(xv + kb);             // A[m][k]=x[k]: broadcast per half
    v2f b = *(const v2f*)(W + row * 16 + kb);   // B[k][n]=W[n][k]: 8B strided,
                                                // 32 lanes cover the full 1KB
    c = __builtin_amdgcn_wmma_f32_16x16x4_f32(
        /*neg_a=*/false, a, /*neg_b=*/false, b,
        /*c_mod=*/(short)0, c, /*reuse_a=*/false, /*reuse_b=*/false);
  }
  return c[0];
#else
  // Fallback (probe-confirmed builtin): one V_WMMA_F32_16X16X32_F16, K padded
  // 16->32 with zeros, same transposed formulation.
  // A f16 16x32: lane<16 -> a[0..7]=K0-7=x[0..7]; lane>=16 -> a[0..7]=K8-15.
  v16h a = {};
  {
    const float4* xp = (const float4*)(xv + half * 8);  // 32B broadcast per half
    float4 t0 = xp[0], t1 = xp[1];
    a[0] = (_Float16)t0.x; a[1] = (_Float16)t0.y;
    a[2] = (_Float16)t0.z; a[3] = (_Float16)t0.w;
    a[4] = (_Float16)t1.x; a[5] = (_Float16)t1.y;
    a[6] = (_Float16)t1.z; a[7] = (_Float16)t1.w;
  }
  // B f16 32x16: lane<16 -> b[j]=W[lane][j], at K=j for j<8, K=8+j for j>=8 (one contiguous
  // 64B row of W per lane); lanes 16-31 are zero, at K8-15 and K24-31.
  v16h b = {};
  if (lane < 16) {
    const float4* wp = (const float4*)(W + row * 16);
#pragma unroll
    for (int v = 0; v < 4; ++v) {
      float4 t = wp[v];
      b[4 * v + 0] = (_Float16)t.x; b[4 * v + 1] = (_Float16)t.y;
      b[4 * v + 2] = (_Float16)t.z; b[4 * v + 3] = (_Float16)t.w;
    }
  }
  v8f c = {};
  c = __builtin_amdgcn_wmma_f32_16x16x32_f16(
      /*neg_a=*/false, a, /*neg_b=*/false, b,
      /*c_mod=*/(short)0, c, /*reuse_a=*/false, /*reuse_b=*/false);
  return c[0];
#endif
}

// out[n] = W_self[n] @ x[n] + b_self[n]   (one node per wave, full overwrite)
__global__ void sheaf_self_kernel(const float* __restrict__ x,
                                  const float* __restrict__ w_self,
                                  const float* __restrict__ b_self,
                                  float* __restrict__ out, int n_nodes) {
  int wave = (int)((blockIdx.x * blockDim.x + threadIdx.x) >> 5);
  int lane = (int)(threadIdx.x & 31);
  if (wave >= n_nodes) return;  // wave-uniform: EXEC all-1s at the WMMA
  size_t n = (size_t)wave;

  float y = matvec16_wmma(w_self + n * 256, x + n * 16, lane);

  if (lane < 16) {  // one 64B row store from 16 lanes (single cacheline)
    out[n * 16 + lane] = y + b_self[n * 16 + lane];
  }
}

// out[dst[e]] += W_edge[e] @ x[src[e]] + b_edge[e]   (one edge per wave)
__global__ void sheaf_edge_kernel(const float* __restrict__ x,
                                  const long long* __restrict__ edge_index,
                                  const float* __restrict__ w_edge,
                                  const float* __restrict__ b_edge,
                                  float* __restrict__ out, int n_edges) {
  int wave = (int)((blockIdx.x * blockDim.x + threadIdx.x) >> 5);
  int lane = (int)(threadIdx.x & 31);
  if (wave >= n_edges) return;  // wave-uniform
  size_t e = (size_t)wave;

  long long src = edge_index[e];
  long long dst = edge_index[(size_t)n_edges + e];

  float y = matvec16_wmma(w_edge + e * 256, x + (size_t)src * 16, lane);

  if (lane < 16) {
    // Single global_atomic_add_f32 instruction: 16 lanes hit one 64B-aligned
    // cacheline of out (L2-resident), instead of 16 serialized lane-atomics.
    float v = y + b_edge[e * 16 + lane];
    unsafeAtomicAdd(out + (size_t)dst * 16 + lane, v);
  }
}

__global__ void sheaf_relu_kernel(float* __restrict__ out, int n) {
  int i = (int)(blockIdx.x * blockDim.x + threadIdx.x);
  if (i < n) out[i] = fmaxf(out[i], 0.0f);
}

extern "C" void kernel_launch(void* const* d_in, const int* in_sizes, int n_in,
                              void* d_out, int out_size, void* d_ws, size_t ws_size,
                              hipStream_t stream) {
  const float*     x      = (const float*)d_in[0];
  const long long* ei     = (const long long*)d_in[1];  // int64 per reference
  const float*     w_edge = (const float*)d_in[2];
  const float*     w_self = (const float*)d_in[3];
  const float*     b_edge = (const float*)d_in[4];
  const float*     b_self = (const float*)d_in[5];
  float*           out    = (float*)d_out;

  const int n_nodes = in_sizes[0] / 16;      // x is (N, 16)
  const int n_edges = in_sizes[1] / 2;       // edge_index is (2, E)

  // 256 threads = 8 waves per block, one node/edge per wave.
  const int blk = 256;
  const int self_blocks = (n_nodes + 7) / 8;
  const int edge_blocks = (n_edges + 7) / 8;
  const int relu_n      = n_nodes * 16;
  const int relu_blocks = (relu_n + blk - 1) / blk;

  sheaf_self_kernel<<<self_blocks, blk, 0, stream>>>(x, w_self, b_self, out, n_nodes);
  sheaf_edge_kernel<<<edge_blocks, blk, 0, stream>>>(x, ei, w_edge, b_edge, out, n_edges);
  sheaf_relu_kernel<<<relu_blocks, blk, 0, stream>>>(out, relu_n);
}
